// LSTMModel_17411797418020
// MI455X (gfx1250) — matrix operand order repaired and checked by reading, not compiled, not run on hardware
//
#include <hip/hip_runtime.h>

typedef __attribute__((ext_vector_type(16))) _Float16 v16h;
typedef __attribute__((ext_vector_type(8)))  float    v8f;

#define DEV __device__ __forceinline__

// ---- WMMA fragment helpers (CDNA5 16x16x32 f16 -> f32) ---------------------
// A (16x32, MxK) per ISA 7.12.2: lanes 0-15 row M=lane, K in {0..7,16..23};
// lanes 16-31 row M=lane-16, K in {8..15,24..31}.
DEV v16h a_frag_lds(const _Float16* tile, int stride, int lane) {
  const _Float16* p = tile + (lane & 15) * stride;
  int cb = (lane >> 4) << 3;   // 0 or 8
  v16h a;
#pragma unroll
  for (int j = 0; j < 8; ++j) a[j] = p[cb + j];
#pragma unroll
  for (int j = 0; j < 8; ++j) a[8 + j] = p[cb + 16 + j];
  return a;
}

// B (32x16, KxN) loaded from B^T stored (N x K) row-major:
// lanes 0-15: col N=lane, K in {0..7,16..23} ; lanes 16-31: col N=lane-16, K in {8..15,24..31}.
DEV v16h b_frag(const _Float16* bt, int stride, int lane) {
  const _Float16* p = bt + (lane & 15) * stride + ((lane >> 4) << 3);
  v16h b;
#pragma unroll
  for (int j = 0; j < 16; ++j) b[j] = p[(j < 8) ? j : (8 + j)];
  return b;
}

DEV v8f wmma_f16(v16h a, v16h b, v8f c) {
  return __builtin_amdgcn_wmma_f32_16x16x32_f16(false, a, false, b,
                                                (short)0, c, false, false);
}

DEV float sigmoidf_(float x) { return 1.0f / (1.0f + __expf(-x)); }

// ---- Async global -> LDS copy (ASYNCcnt path, gfx1250) ---------------------
// VDST = LDS byte offset (low 32 bits of generic shared pointer), VADDR = 64b
// global address. Tracked by ASYNCcnt; waited with s_wait_asynccnt.
DEV void async_ld_b128(const void* lds_dst, const void* gsrc) {
  uint32_t loff = (uint32_t)(uintptr_t)lds_dst;
  asm volatile("global_load_async_to_lds_b128 %0, %1, off"
               :: "v"(loff), "v"(gsrc) : "memory");
}
DEV void async_wait0() { asm volatile("s_wait_asynccnt 0" ::: "memory"); }

// ---------------------------------------------------------------------------
// Kernel: pad/convert Wih (256x316 f32) -> (256x320 f16)
__global__ void k_prep(const float* __restrict__ Wih, _Float16* __restrict__ Wih_h) {
  int idx = blockIdx.x * 256 + threadIdx.x;
  if (idx < 256 * 320) {
    int n = idx / 320, k = idx % 320;
    Wih_h[idx] = (_Float16)((k < 316) ? Wih[n * 316 + k] : 0.f);
  }
}

// ---------------------------------------------------------------------------
// Kernel: feature MLP (960x63 -> 63 -> 64), WMMA, 1 wave per 16 rows.
// Writes xseq[t*16+b][0..63] (seq-order, row stride 320).
__global__ void k_mlp(const float* __restrict__ in, const float* __restrict__ W00,
                      const float* __restrict__ b00, const float* __restrict__ W01,
                      const float* __restrict__ b01, float* __restrict__ xseq) {
  __shared__ _Float16 A1[16 * 64], A2[16 * 64], W0[64 * 64], W1[64 * 64];
  int tid = threadIdx.x, lane = tid & 31;
  int r0 = blockIdx.x * 16;
  for (int idx = tid; idx < 16 * 64; idx += 32) {
    int m = idx >> 6, k = idx & 63;
    A1[idx] = (_Float16)((k < 63) ? in[(r0 + m) * 63 + k] : 0.f);
  }
  for (int idx = tid; idx < 64 * 64; idx += 32) {
    int n = idx >> 6, k = idx & 63;
    W0[idx] = (_Float16)((n < 63 && k < 63) ? W00[n * 63 + k] : 0.f);
    W1[idx] = (_Float16)((k < 63) ? W01[n * 63 + k] : 0.f);
  }
  __syncthreads();
#pragma unroll
  for (int nt = 0; nt < 4; ++nt) {            // layer 1 -> A2 (relu)
    v8f acc = {};
#pragma unroll
    for (int kc = 0; kc < 2; ++kc)
      acc = wmma_f16(a_frag_lds(A1 + kc * 32, 64, lane),
                     b_frag(W0 + nt * 16 * 64 + kc * 32, 64, lane), acc);
    int n = nt * 16 + (lane & 15);
    float bias = (n < 63) ? b00[n] : 0.f;
#pragma unroll
    for (int v = 0; v < 8; ++v) {
      int m = v + ((lane >> 4) << 3);
      A2[m * 64 + n] = (_Float16)fmaxf(acc[v] + bias, 0.f);
    }
  }
  __syncthreads();
#pragma unroll
  for (int nt = 0; nt < 4; ++nt) {            // layer 2 -> xseq (relu)
    v8f acc = {};
#pragma unroll
    for (int kc = 0; kc < 2; ++kc)
      acc = wmma_f16(a_frag_lds(A2 + kc * 32, 64, lane),
                     b_frag(W1 + nt * 16 * 64 + kc * 32, 64, lane), acc);
    int n = nt * 16 + (lane & 15);
    float bias = b01[n];
#pragma unroll
    for (int v = 0; v < 8; ++v) {
      int m = v + ((lane >> 4) << 3);
      int g = r0 + m, bb = g / 60, tt = g % 60;
      xseq[(tt * 16 + bb) * 320 + n] = fmaxf(acc[v] + bias, 0.f);
    }
  }
}

// ---------------------------------------------------------------------------
// Kernel: fused conv0 (1->16ch, 5x5 s4) + conv1 (16->16ch, 5x5 s4), WMMA.
// Block = (oy1, img): computes one conv1 output row (19 px) incl. the 5 conv0
// rows it needs; conv0 results staged in LDS as f16. Video staged via async
// global->LDS (b128, ASYNCcnt). Conv1 K-loop split 4-ways per M-tile with an
// LDS partial reduction so all 8 waves stay busy. out2: (img,16,14,19) f32.
__global__ void k_conv01(const float* __restrict__ video, const float* __restrict__ Wc0,
                         const float* __restrict__ bc0, const float* __restrict__ Wc1,
                         const float* __restrict__ bc1, float* __restrict__ out2) {
  __shared__ __align__(16) float vid[21 * 320];
  __shared__ _Float16 c0o[16 * 5 * 79];   // [oc][r][ox]
  __shared__ _Float16 wc0[16 * 32];
  __shared__ _Float16 wc1[16 * 416];
  __shared__ float    psum[8 * 256];      // [tile*4+part][m*16+oc]
  int tid = threadIdx.x, lane = tid & 31, wave = tid >> 5;
  int oy1 = blockIdx.x, img = blockIdx.y;
  // async-stage 21 video rows (1680 x b128) while converting weights to f16
  const float* vsrc = video + (size_t)img * 240 * 320 + (size_t)(16 * oy1) * 320;
  for (int idx = tid; idx < 1680; idx += 256)
    async_ld_b128(vid + idx * 4, vsrc + idx * 4);
  for (int idx = tid; idx < 16 * 32; idx += 256) {
    int oc = idx >> 5, k = idx & 31;
    wc0[idx] = (_Float16)((k < 25) ? Wc0[oc * 25 + k] : 0.f);
  }
  for (int idx = tid; idx < 16 * 416; idx += 256) {
    int oc = idx / 416, k = idx % 416;
    wc1[idx] = (_Float16)((k < 400) ? Wc1[oc * 400 + k] : 0.f);
  }
  async_wait0();
  __syncthreads();
  // conv0: 25 jobs = (r 0..4) x (ox-tile 0..4), K=25 padded to 32 (1 WMMA each)
  v16h b0 = b_frag(wc0, 32, lane);
  for (int job = wave; job < 25; job += 8) {
    int r = job / 5, tix = job % 5;
    int ox = tix * 16 + (lane & 15);
    int oxc = (ox < 79) ? ox : 78;
    int cb = (lane >> 4) << 3;
    v16h a;
#pragma unroll
    for (int j = 0; j < 16; ++j) {
      int c = cb + ((j < 8) ? j : (8 + j));
      float val = 0.f;
      if (c < 25) { int dy = c / 5, dx = c % 5; val = vid[(4 * r + dy) * 320 + oxc * 4 + dx]; }
      a[j] = (_Float16)val;
    }
    v8f acc = {};
    acc = wmma_f16(a, b0, acc);
    int oc = lane & 15;
    float bias = bc0[oc];
#pragma unroll
    for (int v = 0; v < 8; ++v) {
      int oxv = tix * 16 + v + ((lane >> 4) << 3);
      if (oxv < 79)
        c0o[oc * 395 + r * 79 + oxv] = (_Float16)fmaxf(acc[v] + bias, 0.f);
    }
  }
  __syncthreads();
  // conv1: 19 px = 2 M-tiles; K=400->416 = 13 k-chunks split over 4 waves/tile
  {
    int t = wave >> 2;            // m-tile (0,1)
    int part = wave & 3;          // k-split (0..3)
    int p = t * 16 + (lane & 15);
    int pc = (p < 19) ? p : 18;
    int cb = (lane >> 4) << 3;
    v8f acc = {};
    for (int kc = part; kc < 13; kc += 4) {
      v16h a;
#pragma unroll
      for (int j = 0; j < 16; ++j) {
        int c = kc * 32 + cb + ((j < 8) ? j : (8 + j));
        float val = 0.f;
        if (c < 400) {
          int ic = c / 25, rr = c % 25, dy = rr / 5, dx = rr % 5;
          val = (float)c0o[ic * 395 + dy * 79 + 4 * pc + dx];
        }
        a[j] = (_Float16)val;
      }
      acc = wmma_f16(a, b_frag(wc1 + kc * 32, 416, lane), acc);
    }
#pragma unroll
    for (int v = 0; v < 8; ++v) {
      int m = v + ((lane >> 4) << 3);
      psum[wave * 256 + m * 16 + (lane & 15)] = acc[v];
    }
  }
  __syncthreads();
  // reduce 4 partials per tile, bias + relu, store
  for (int slot = tid; slot < 512; slot += 256) {
    int t = slot >> 8, e = slot & 255;
    int m = e >> 4, oc = e & 15;
    int p = t * 16 + m;
    if (p < 19) {
      int b0i = t * 4;
      float s = psum[(b0i + 0) * 256 + e] + psum[(b0i + 1) * 256 + e] +
                psum[(b0i + 2) * 256 + e] + psum[(b0i + 3) * 256 + e];
      out2[((size_t)img * 16 + oc) * 266 + oy1 * 19 + p] = fmaxf(s + bc1[oc], 0.f);
    }
  }
}

// ---------------------------------------------------------------------------
// Kernel: conv2 (16->4ch, 2x2 s2) + write into xseq cols [64,316) + zero pad.
// Activations staged via async global->LDS.
__global__ void k_conv2(const float* __restrict__ out2, const float* __restrict__ Wc2,
                        const float* __restrict__ bc2, float* __restrict__ xseq) {
  __shared__ __align__(16) float in2[16 * 266];
  __shared__ float w2[256];
  int tid = threadIdx.x, img = blockIdx.x;
  const float* src = out2 + (size_t)img * 16 * 266;
  for (int idx = tid; idx < 1064; idx += 256)          // 4256 dwords / 4
    async_ld_b128(in2 + idx * 4, src + idx * 4);
  w2[tid] = Wc2[tid];
  async_wait0();
  __syncthreads();
  int bb = img / 60, tt = img % 60, srow = tt * 16 + bb;
  if (tid < 252) {
    int oc = tid / 63, rem = tid % 63, oy = rem / 9, ox = rem % 9;
    float acc = bc2[oc];
    for (int ic = 0; ic < 16; ++ic)
#pragma unroll
      for (int dy = 0; dy < 2; ++dy)
#pragma unroll
        for (int dx = 0; dx < 2; ++dx)
          acc += in2[ic * 266 + (2 * oy + dy) * 19 + (2 * ox + dx)] *
                 w2[oc * 64 + ic * 4 + dy * 2 + dx];
    xseq[srow * 320 + 64 + tid] = fmaxf(acc, 0.f);
  } else {
    xseq[srow * 320 + 64 + tid] = 0.f;  // cols 316..319 padding
  }
}

// ---------------------------------------------------------------------------
// Kernel: G = xseq @ Wih^T + bih + bhh  (960x320 f16 @ 320x256 -> f32), WMMA.
__global__ void k_gemmG(const float* __restrict__ xseq, const _Float16* __restrict__ Wih_h,
                        const float* __restrict__ bih, const float* __restrict__ bhh,
                        float* __restrict__ G) {
  __shared__ _Float16 Ah[16 * 320];
  int tid = threadIdx.x, lane = tid & 31, wave = tid >> 5;
  int s0 = blockIdx.x * 16;
  for (int idx = tid; idx < 16 * 320; idx += 256)
    Ah[idx] = (_Float16)xseq[(size_t)s0 * 320 + idx];
  __syncthreads();
#pragma unroll
  for (int q = 0; q < 2; ++q) {
    int nt = wave * 2 + q;
    v8f acc = {};
    for (int kc = 0; kc < 10; ++kc) {
      const _Float16* bp = Wih_h + (size_t)(nt * 16 + (lane & 15)) * 320 +
                           kc * 32 + ((lane >> 4) << 3);
      if (kc < 9) __builtin_prefetch(bp + 32, 0, 1);
      v16h b;
#pragma unroll
      for (int j = 0; j < 16; ++j) b[j] = bp[(j < 8) ? j : (8 + j)];
      acc = wmma_f16(a_frag_lds(Ah + kc * 32, 320, lane), b, acc);
    }
    int n = nt * 16 + (lane & 15);
    float bias = bih[n] + bhh[n];
#pragma unroll
    for (int v = 0; v < 8; ++v) {
      int m = v + ((lane >> 4) << 3);
      G[(size_t)(s0 + m) * 256 + n] = acc[v] + bias;
    }
  }
}

// ---------------------------------------------------------------------------
// Kernel: sequential LSTM scan, 960 steps, single block. Whh row per thread in
// VGPRs; h, gates in LDS. Stores last 16 hidden states.
__global__ void k_lstm(const float* __restrict__ G, const float* __restrict__ Whh,
                       float* __restrict__ hlast) {
  __shared__ float hlds[64];
  __shared__ float gbuf[256];
  int j = threadIdx.x;
  float w[64];
#pragma unroll
  for (int k = 0; k < 64; ++k) w[k] = Whh[j * 64 + k];
  if (j < 64) hlds[j] = 0.f;
  float c = 0.f;
  __syncthreads();
  for (int t = 0; t < 960; ++t) {
    float acc = G[t * 256 + j];
#pragma unroll
    for (int k = 0; k < 64; ++k) acc += hlds[k] * w[k];
    gbuf[j] = acc;
    __syncthreads();
    if (j < 64) {
      float ig = gbuf[j], fg = gbuf[64 + j], gg = gbuf[128 + j], og = gbuf[192 + j];
      c = sigmoidf_(fg) * c + sigmoidf_(ig) * tanhf(gg);
      float h = sigmoidf_(og) * tanhf(c);
      hlds[j] = h;
      if (t >= 944) hlast[(t - 944) * 64 + j] = h;
    }
    __syncthreads();
  }
}

// ---------------------------------------------------------------------------
// Kernel: head — relu(h@W10^T+b10) -> relu(@W11^T+b11) -> @W12^T+b12.
__global__ void k_head(const float* __restrict__ hl, const float* __restrict__ W10,
                       const float* __restrict__ b10, const float* __restrict__ W11,
                       const float* __restrict__ b11, const float* __restrict__ W12,
                       const float* __restrict__ b12, float* __restrict__ out) {
  __shared__ float h0[16 * 64], o1[16 * 32], o2[16 * 32];
  int tid = threadIdx.x;  // 512
  for (int idx = tid; idx < 16 * 64; idx += 512) h0[idx] = hl[idx];
  __syncthreads();
  {
    int b = tid >> 5, n = tid & 31;
    float acc = b10[n];
    for (int k = 0; k < 64; ++k) acc += h0[b * 64 + k] * W10[n * 64 + k];
    o1[b * 32 + n] = fmaxf(acc, 0.f);
  }
  __syncthreads();
  {
    int b = tid >> 5, n = tid & 31;
    float acc = b11[n];
    for (int k = 0; k < 32; ++k) acc += o1[b * 32 + k] * W11[n * 32 + k];
    o2[b * 32 + n] = fmaxf(acc, 0.f);
  }
  __syncthreads();
  if (tid < 256) {
    int n = tid;
    float wr[32];
#pragma unroll
    for (int k = 0; k < 32; ++k) wr[k] = W12[n * 32 + k];
    for (int b = 0; b < 16; ++b) {
      float acc = b12[n];
#pragma unroll
      for (int k = 0; k < 32; ++k) acc += o2[b * 32 + k] * wr[k];
      out[b * 256 + n] = acc;
    }
  }
}

// ---------------------------------------------------------------------------
extern "C" void kernel_launch(void* const* d_in, const int* in_sizes, int n_in,
                              void* d_out, int out_size, void* d_ws, size_t ws_size,
                              hipStream_t stream) {
  (void)in_sizes; (void)n_in; (void)out_size; (void)ws_size;
  const float* input = (const float*)d_in[0];
  const float* video = (const float*)d_in[1];
  const float* W00 = (const float*)d_in[2];
  const float* b00 = (const float*)d_in[3];
  const float* W01 = (const float*)d_in[4];
  const float* b01 = (const float*)d_in[5];
  const float* Wc0 = (const float*)d_in[6];
  const float* bc0 = (const float*)d_in[7];
  const float* Wc1 = (const float*)d_in[8];
  const float* bc1 = (const float*)d_in[9];
  const float* Wc2 = (const float*)d_in[10];
  const float* bc2 = (const float*)d_in[11];
  const float* Wih = (const float*)d_in[12];
  const float* Whh = (const float*)d_in[13];
  const float* bih = (const float*)d_in[14];
  const float* bhh = (const float*)d_in[15];
  const float* W10 = (const float*)d_in[16];
  const float* b10 = (const float*)d_in[17];
  const float* W11 = (const float*)d_in[18];
  const float* b11 = (const float*)d_in[19];
  const float* W12 = (const float*)d_in[20];
  const float* b12 = (const float*)d_in[21];
  float* out = (float*)d_out;

  // workspace layout (total ~18.7 MB)
  char* ws = (char*)d_ws;
  _Float16* Wih_h = (_Float16*)ws;                              // 256*320*2   = 163,840 B
  float* out2 = (float*)(ws + 163840);                          // 960*16*266*4 = 16,343,040 B
  float* xseq = (float*)(ws + 163840 + 16343040);               // 960*320*4   = 1,228,800 B
  float* G    = (float*)(ws + 163840 + 16343040 + 1228800);     // 960*256*4   = 983,040 B
  float* hl   = (float*)(ws + 163840 + 16343040 + 1228800 + 983040);  // 16*64*4

  k_prep  <<<320, 256, 0, stream>>>(Wih, Wih_h);
  k_mlp   <<<60, 32, 0, stream>>>(input, W00, b00, W01, b01, xseq);
  k_conv01<<<dim3(14, 960), 256, 0, stream>>>(video, Wc0, bc0, Wc1, bc1, out2);
  k_conv2 <<<960, 256, 0, stream>>>(out2, Wc2, bc2, xseq);
  k_gemmG <<<60, 256, 0, stream>>>(xseq, Wih_h, bih, bhh, G);
  k_lstm  <<<1, 256, 0, stream>>>(G, Whh, hl);
  k_head  <<<1, 512, 0, stream>>>(hl, W10, b10, W11, b11, W12, b12, out);
}
